// WindowedCrossAttn_16518444220994
// MI455X (gfx1250) — compile-verified, matrix operand order checked by reading, not run on hardware
//
#include <hip/hip_runtime.h>
#include <stdint.h>
#include <stddef.h>

#define B_ 2
#define S_ 4096
#define D_ 1024
#define H_ 16
#define DH_ 64
#define WIN_ 256
#define M_ (B_ * S_) /* 8192 rows for the projection GEMMs */

static constexpr float SCALE_ = 0.125f; /* 1/sqrt(64) */

typedef __attribute__((ext_vector_type(16))) __bf16 v16bf;
typedef __attribute__((ext_vector_type(8)))  float  v8f;

__device__ __forceinline__ uint16_t f2bf(float f) {
  union { float f; uint32_t u; } cv; cv.f = f;
  uint32_t u = cv.u;
  u += 0x7FFFu + ((u >> 16) & 1u);   /* round-to-nearest-even */
  return (uint16_t)(u >> 16);
}

/* Build a 16-half bf16 fragment from two contiguous 16-byte chunks.
 * Wave32 layout: lane L holds row (L&15); halves[0..8) = K ksel..ksel+8,
 * halves[8..16) = K ksel+16..ksel+24, ksel = 8*(L>>4). */
__device__ __forceinline__ v16bf load_frag(const uint16_t* p0, const uint16_t* p1) {
  union { v16bf v; uint4 q[2]; } u;
  u.q[0] = *(const uint4*)p0;
  u.q[1] = *(const uint4*)p1;
  return u.v;
}

/* ---------------- f32 -> bf16 conversion ---------------- */
__global__ void cvt_f32_bf16(const float* __restrict__ src,
                             uint16_t* __restrict__ dst, int n) {
  int i = blockIdx.x * blockDim.x + threadIdx.x;
  int stride = gridDim.x * blockDim.x;
  for (; i < n; i += stride) dst[i] = f2bf(src[i]);
}

/* ---------------- bf16 GEMM:  out[M,N] = A[M,K] @ W[N,K]^T + bias ----------
 * mode 0: out bf16 as [B,H,S,64]     (Q, K projections)
 * mode 1: out bf16 as [B,H,64,S]     (V projection, transposed per head)
 * mode 2: out f32 row-major [M,N]    (final Wo projection)
 * Each wave computes a 32x64 tile (2 M-subtiles x 4 N-chunks, 8 f32
 * accumulators): per K-step, 2 A-frags are crossed with 4 B-frags -> 8 WMMAs
 * from 12 x 16B loads, and the shared W fragments amortize over 2 M-rows.
 * Block = 8 waves stacked in M (256x64 block tile).  Grid: (M/256, N/64).   */
__global__ __launch_bounds__(256) void gemm_bf16(
    const uint16_t* __restrict__ A, const uint16_t* __restrict__ W,
    const float* __restrict__ bias, uint16_t* __restrict__ out_bf,
    float* __restrict__ out_f32, int mode) {
  const int lane = threadIdx.x & 31;
  const int wave = threadIdx.x >> 5;
  const int r0 = (blockIdx.x * 8 + wave) * 32;
  const int c0 = blockIdx.y * 64;
  const int lm = lane & 15;
  const int half = lane >> 4;
  const int ksel = half << 3;

  v8f acc[2][4] = {};
  const uint16_t* arow0 = A + (size_t)(r0 + lm) * D_;
  const uint16_t* arow1 = arow0 + (size_t)16 * D_;
  const uint16_t* wbase = W + (size_t)(c0 + lm) * D_;

  for (int k = 0; k < D_; k += 32) {
    /* prefetch the A streams a few K-steps ahead (global_prefetch_b8) */
    __builtin_prefetch(arow0 + k + 256, 0, 1);
    __builtin_prefetch(arow1 + k + 256, 0, 1);
    v16bf a0 = load_frag(arow0 + k + ksel, arow0 + k + ksel + 16);
    v16bf a1 = load_frag(arow1 + k + ksel, arow1 + k + ksel + 16);
#pragma unroll
    for (int c = 0; c < 4; ++c) {
      const uint16_t* wrow = wbase + (size_t)(c * 16) * D_;
      v16bf b = load_frag(wrow + k + ksel, wrow + k + ksel + 16);
      acc[0][c] = __builtin_amdgcn_wmma_f32_16x16x32_bf16(
          false, a0, false, b, (short)0, acc[0][c], false, false);
      acc[1][c] = __builtin_amdgcn_wmma_f32_16x16x32_bf16(
          false, a1, false, b, (short)0, acc[1][c], false, false);
    }
  }

#pragma unroll
  for (int ms = 0; ms < 2; ++ms)
#pragma unroll
    for (int c = 0; c < 4; ++c) {
      const int n_abs = c0 + c * 16 + lm;
      const float bv = bias[n_abs];
#pragma unroll
      for (int jv = 0; jv < 8; ++jv) {
        const int m_abs = r0 + ms * 16 + jv + 8 * half;
        const float v = acc[ms][c][jv] + bv;
        if (mode == 2) {
          out_f32[(size_t)m_abs * D_ + n_abs] = v;
        } else {
          const int b = m_abs >> 12, s = m_abs & (S_ - 1);
          const int h = n_abs >> 6, d = n_abs & (DH_ - 1);
          if (mode == 0)
            out_bf[(((size_t)(b * H_ + h)) * S_ + s) * DH_ + d] = f2bf(v);
          else /* transposed per head: [B,H,64,S] */
            out_bf[(((size_t)(b * H_ + h)) * DH_ + d) * S_ + s] = f2bf(v);
        }
      }
    }
}

/* ---------------- windowed flash attention ----------------
 * One wave per 16-query tile.  Streams the causal 256-wide band in 32-key
 * chunks: 4 WMMAs Q@K^T, online softmax, P via LDS (C-layout -> A-layout),
 * 4 WMMAs P@V into f32 accumulators.                                        */
__global__ __launch_bounds__(256) void attn_win(
    const uint16_t* __restrict__ Qh,  /* [B*H, S, 64]  */
    const uint16_t* __restrict__ Kh,  /* [B*H, S, 64]  */
    const uint16_t* __restrict__ VT,  /* [B*H, 64, S]  */
    uint16_t* __restrict__ ctx) {     /* [B, S, 1024]  */
  __shared__ uint16_t pbuf[8][16 * 32]; /* per-wave 16x32 bf16 P tile */
  const int lane = threadIdx.x & 31;
  const int wave = threadIdx.x >> 5;
  const int t = blockIdx.x * 8 + wave;   /* 8192 q-tiles total */
  const int bh = t >> 8;                 /* 256 tiles per (b,h) */
  const int i0 = (t & 255) * 16;
  const int lm = lane & 15;
  const int half = lane >> 4;
  const int ksel = half << 3;

  const uint16_t* Qb = Qh + (size_t)bh * S_ * DH_;
  const uint16_t* Kb = Kh + (size_t)bh * S_ * DH_;
  const uint16_t* Vb = VT + (size_t)bh * DH_ * S_;

  const uint16_t* qrow = Qb + (size_t)(i0 + lm) * DH_;
  const v16bf aq0 = load_frag(qrow + ksel,      qrow + ksel + 16);
  const v16bf aq1 = load_frag(qrow + 32 + ksel, qrow + 32 + ksel + 16);

  v8f acc[4] = {};
  float mi[8], li[8];
#pragma unroll
  for (int jv = 0; jv < 8; ++jv) { mi[jv] = -3.0e38f; li[jv] = 0.0f; }

  int lo = i0 - (WIN_ - 1); if (lo < 0) lo = 0;
  const int jstart = lo & ~31;
  const int niter = ((i0 + 15 - jstart) >> 5) + 1;

  for (int it = 0; it < niter; ++it) {
    const int j0 = jstart + 32 * it;
    v8f st[2];
#pragma unroll
    for (int tile = 0; tile < 2; ++tile) {
      int jr = j0 + 16 * tile + lm;
      if (jr > S_ - 1) jr = S_ - 1;  /* clamp; masked below anyway */
      const uint16_t* krow = Kb + (size_t)jr * DH_;
      const v16bf bk0 = load_frag(krow + ksel,      krow + ksel + 16);
      const v16bf bk1 = load_frag(krow + 32 + ksel, krow + 32 + ksel + 16);
      v8f s = {};
      s = __builtin_amdgcn_wmma_f32_16x16x32_bf16(false, aq0, false, bk0,
                                                  (short)0, s, false, false);
      s = __builtin_amdgcn_wmma_f32_16x16x32_bf16(false, aq1, false, bk1,
                                                  (short)0, s, false, false);
      st[tile] = s;
    }

    /* scale + band mask + online softmax (rows live in 16-lane halves) */
#pragma unroll
    for (int jv = 0; jv < 8; ++jv) {
      const int q_abs = i0 + jv + 8 * half;
#pragma unroll
      for (int tile = 0; tile < 2; ++tile) {
        const int j_abs = j0 + 16 * tile + lm;
        const bool ok = (j_abs <= q_abs) && (j_abs + WIN_ > q_abs);
        const float sv = st[tile][jv];
        st[tile][jv] = ok ? sv * SCALE_ : -3.0e38f;
      }
      float r = fmaxf(st[0][jv], st[1][jv]);
#pragma unroll
      for (int off = 1; off < 16; off <<= 1)
        r = fmaxf(r, __shfl_xor(r, off, 32));
      const float mnew = fmaxf(mi[jv], r);
      const float alpha = __expf(mi[jv] - mnew);
      mi[jv] = mnew;
      const float p0 = __expf(st[0][jv] - mnew);
      const float p1 = __expf(st[1][jv] - mnew);
      st[0][jv] = p0; st[1][jv] = p1;
      float rs = p0 + p1;
#pragma unroll
      for (int off = 1; off < 16; off <<= 1)
        rs += __shfl_xor(rs, off, 32);
      li[jv] = li[jv] * alpha + rs;
#pragma unroll
      for (int c = 0; c < 4; ++c) acc[c][jv] *= alpha;
    }

    /* P: C-layout -> LDS -> A-layout (same wave; dscnt ordering by compiler) */
#pragma unroll
    for (int tile = 0; tile < 2; ++tile)
#pragma unroll
      for (int jv = 0; jv < 8; ++jv) {
        const int m = jv + 8 * half;
        pbuf[wave][m * 32 + 16 * tile + lm] = f2bf(st[tile][jv]);
      }
    const uint16_t* prow = &pbuf[wave][lm * 32];
    const v16bf ap = load_frag(prow + ksel, prow + ksel + 16);

    /* P @ V: B-fragments contiguous thanks to transposed VT rows */
#pragma unroll
    for (int c = 0; c < 4; ++c) {
      const uint16_t* vrow = Vb + (size_t)(c * 16 + lm) * S_ + j0;
      const v16bf bvf = load_frag(vrow + ksel, vrow + ksel + 16);
      acc[c] = __builtin_amdgcn_wmma_f32_16x16x32_bf16(
          false, ap, false, bvf, (short)0, acc[c], false, false);
    }
  }

  /* normalize and scatter to ctx[B,S,1024] */
  const int b = bh >> 4, h = bh & 15;
#pragma unroll
  for (int c = 0; c < 4; ++c)
#pragma unroll
    for (int jv = 0; jv < 8; ++jv) {
      const int s = i0 + jv + 8 * half;
      const int d = c * 16 + lm;
      const float v = acc[c][jv] / li[jv];
      ctx[((size_t)(b * S_ + s)) * D_ + h * DH_ + d] = f2bf(v);
    }
}

extern "C" void kernel_launch(void* const* d_in, const int* in_sizes, int n_in,
                              void* d_out, int out_size, void* d_ws, size_t ws_size,
                              hipStream_t stream) {
  const float* q_in = (const float*)d_in[0];
  const float* k_in = (const float*)d_in[1];
  const float* v_in = (const float*)d_in[2];
  const float* Wq = (const float*)d_in[3];
  const float* bq = (const float*)d_in[4];
  const float* Wk = (const float*)d_in[5];
  const float* bk = (const float*)d_in[6];
  const float* Wv = (const float*)d_in[7];
  const float* bv = (const float*)d_in[8];
  const float* Wo = (const float*)d_in[9];
  const float* bo = (const float*)d_in[10];

  const size_t IN = (size_t)M_ * D_;  /* 8,388,608 elements */
  const size_t WE = (size_t)D_ * D_;  /* 1,048,576 elements */

  uint16_t* ws = (uint16_t*)d_ws;
  uint16_t* xq = ws;            /* bf16 inputs  */
  uint16_t* xk = xq + IN;
  uint16_t* xv = xk + IN;
  uint16_t* wq = xv + IN;       /* bf16 weights */
  uint16_t* wk = wq + WE;
  uint16_t* wv = wk + WE;
  uint16_t* wo = wv + WE;
  uint16_t* Qh = wo + WE;       /* [B,H,S,64]   */
  uint16_t* Kh = Qh + IN;
  uint16_t* Vt = Kh + IN;       /* [B,H,64,S]   */
  uint16_t* ctx = Vt + IN;      /* [B,S,1024]   */

  cvt_f32_bf16<<<2048, 256, 0, stream>>>(q_in, xq, (int)IN);
  cvt_f32_bf16<<<2048, 256, 0, stream>>>(k_in, xk, (int)IN);
  cvt_f32_bf16<<<2048, 256, 0, stream>>>(v_in, xv, (int)IN);
  cvt_f32_bf16<<<512, 256, 0, stream>>>(Wq, wq, (int)WE);
  cvt_f32_bf16<<<512, 256, 0, stream>>>(Wk, wk, (int)WE);
  cvt_f32_bf16<<<512, 256, 0, stream>>>(Wv, wv, (int)WE);
  cvt_f32_bf16<<<512, 256, 0, stream>>>(Wo, wo, (int)WE);

  dim3 gg(M_ / 256, D_ / 64), gb(256);
  gemm_bf16<<<gg, gb, 0, stream>>>(xq, wq, bq, Qh, nullptr, 0);
  gemm_bf16<<<gg, gb, 0, stream>>>(xk, wk, bk, Kh, nullptr, 0);
  gemm_bf16<<<gg, gb, 0, stream>>>(xv, wv, bv, Vt, nullptr, 1);

  attn_win<<<(B_ * H_ * (S_ / 16)) / 8, 256, 0, stream>>>(Qh, Kh, Vt, ctx);

  gemm_bf16<<<gg, gb, 0, stream>>>(ctx, wo, bo, nullptr, (float*)d_out, 2);
}
